// LSTM_18296560681434
// MI455X (gfx1250) — compile-verified, matrix operand order checked by reading, not run on hardware
//
#include <hip/hip_runtime.h>
#include <hip/hip_fp16.h>
#include <math.h>

typedef __attribute__((ext_vector_type(16))) _Float16 v16h;
typedef __attribute__((ext_vector_type(8)))  _Float16 v8h;
typedef __attribute__((ext_vector_type(8)))  float    v8f;

#define B_   128
#define T_   256
#define I_   256
#define H_   512
#define O_   128
#define G4H  (4 * H_)   // 2048

#define WMMA_F16(A, Bv, C) \
    __builtin_amdgcn_wmma_f32_16x16x32_f16(false, (A), false, (Bv), (short)0, (C), false, false)

__device__ __forceinline__ v16h load_frag(const _Float16* p) {
    v16h r;
    ((v8h*)&r)[0] = *(const v8h*)(p);
    ((v8h*)&r)[1] = *(const v8h*)(p + 16);
    return r;
}

// ---------------------------------------------------------------------------
// f32 -> f16 conversion (weights + activations)
// ---------------------------------------------------------------------------
__global__ void cvt_f16_kernel(const float* __restrict__ src,
                               _Float16* __restrict__ dst, int n) {
    int i = blockIdx.x * blockDim.x + threadIdx.x;
    if (i < n) dst[i] = (_Float16)src[i];
}

// Zero initial hidden state (f16) and cell state (f32)
__global__ void init_state_kernel(_Float16* __restrict__ h0,
                                  float* __restrict__ c0, int n) {
    int i = blockIdx.x * blockDim.x + threadIdx.x;
    if (i < n) { h0[i] = (_Float16)0.0f; c0[i] = 0.0f; }
}

// ---------------------------------------------------------------------------
// Input projection GEMM, register-blocked 1x4:
//   out[M,N] (f16) = A[M,K] (f16) @ W[N,K]^T + b0 + b1
// One wave computes a 16x64 output strip: 1 shared A-frag, 4 B-frags,
// 4 independent accumulators -> 160 B of loads per WMMA instead of 256 B,
// and 4-way independent WMMA issue (no single dependent chain).
// Fragment layout (wave32):
//   A lane L: row m0+(L&15), halves at K = (L>>4)*8 .. +7 and +16 .. +23
//   B lane L: col n0+(L&15) = W row, same K pattern (W is [N,K] row-major)
//   C/D lane L: col n0+(L&15), rows m0+(L>>4)*8 + j
// ---------------------------------------------------------------------------
__global__ void gemm_proj_kernel(const _Float16* __restrict__ A,
                                 const _Float16* __restrict__ W,
                                 const float* __restrict__ bias0,
                                 const float* __restrict__ bias1,
                                 _Float16* __restrict__ out,
                                 int M, int N, int K) {
    int wave = blockIdx.x * (blockDim.x >> 5) + (threadIdx.x >> 5);
    int lane = threadIdx.x & 31;
    int ng   = N >> 6;                    // groups of 4 n-tiles (64 cols)
    int m0   = (wave / ng) << 4;
    int n0   = (wave % ng) << 6;
    int lr   = lane & 15;
    int kh   = (lane >> 4) << 3;          // 0 or 8

    const _Float16* ap = A + (size_t)(m0 + lr) * K + kh;
    const _Float16* bp = W + (size_t)(n0 + lr) * K + kh;
    const size_t bstep = (size_t)16 * K;  // next n-tile = +16 W rows

    v8f acc0 = {}, acc1 = {}, acc2 = {}, acc3 = {};
    for (int k = 0; k < K; k += 32) {
        v16h a  = load_frag(ap + k);
        v16h b0 = load_frag(bp + k);
        v16h b1 = load_frag(bp + k + bstep);
        v16h b2 = load_frag(bp + k + 2 * bstep);
        v16h b3 = load_frag(bp + k + 3 * bstep);
        acc0 = WMMA_F16(a, b0, acc0);
        acc1 = WMMA_F16(a, b1, acc1);
        acc2 = WMMA_F16(a, b2, acc2);
        acc3 = WMMA_F16(a, b3, acc3);
    }

    int rbase = m0 + (lane >> 4) * 8;
#define PROJ_EPILOGUE(ACC, U)                                             \
    {                                                                     \
        int col    = n0 + (U) * 16 + lr;                                  \
        float bias = bias0[col] + bias1[col];                             \
        _Float16* op = out + (size_t)rbase * N + col;                     \
        _Pragma("unroll")                                                 \
        for (int j = 0; j < 8; ++j)                                       \
            op[(size_t)j * N] = (_Float16)((ACC)[j] + bias);              \
    }
    PROJ_EPILOGUE(acc0, 0)
    PROJ_EPILOGUE(acc1, 1)
    PROJ_EPILOGUE(acc2, 2)
    PROJ_EPILOGUE(acc3, 3)
#undef PROJ_EPILOGUE
}

// ---------------------------------------------------------------------------
// Fused LSTM step: one block = 16(batch) x 32(hidden) patch, 4 waves.
// Wave g computes gate g for both 16-col n-tiles (2 accumulators sharing
// the h fragment -> two interleaved independent WMMA chains).
// Gate tiles exchanged via LDS, then cell update done cooperatively.
// xp layout: [B, T, 4H] f16 (biases folded in). Whh: [4H, H] f16.
// ---------------------------------------------------------------------------
__global__ void lstm_step_kernel(const _Float16* __restrict__ xp,
                                 const _Float16* __restrict__ Whh,
                                 const _Float16* __restrict__ hin,   // [B,H] f16
                                 _Float16* __restrict__ hout,        // [B,H] f16
                                 float* __restrict__ cbuf,           // [B,H] f32
                                 _Float16* __restrict__ hseq,        // [B,T,H] or null
                                 int t) {
    __shared__ float lds[4][16][33];

    int gate = threadIdx.x >> 5;           // wave id == gate (i,f,g,o)
    int lane = threadIdx.x & 31;
    int m0   = (blockIdx.x >> 4) << 4;     // batch rows   (B/16  = 8)
    int n0   = (blockIdx.x & 15) << 5;     // hidden cols  (H/32  = 16)
    int lr   = lane & 15;
    int kh   = (lane >> 4) << 3;
    int rbase = m0 + (lane >> 4) * 8;

    // Seed accumulators with precomputed input projection (biases folded in)
    v8f acc0, acc1;
    {
        const _Float16* xpp =
            xp + ((size_t)rbase * T_ + t) * G4H + gate * H_ + n0 + lr;
#pragma unroll
        for (int j = 0; j < 8; ++j) {
            size_t o = (size_t)j * T_ * G4H;
            acc0[j] = (float)xpp[o];
            acc1[j] = (float)xpp[o + 16];
        }
        // Warm L2/L0 for the next timestep's xp slice (global_prefetch_b8)
        if (t + 1 < T_) {
            __builtin_prefetch(xpp + G4H, 0, 1);
            __builtin_prefetch(xpp + (size_t)4 * T_ * G4H + G4H, 0, 1);
        }
    }

    const _Float16* ap  = hin + (size_t)(m0 + lr) * H_ + kh;
    const _Float16* bp0 = Whh + (size_t)(gate * H_ + n0 + lr) * H_ + kh;
    const _Float16* bp1 = bp0 + (size_t)16 * H_;
#pragma unroll 4
    for (int k = 0; k < H_; k += 32) {
        v16h a  = load_frag(ap  + k);
        v16h b0 = load_frag(bp0 + k);
        v16h b1 = load_frag(bp1 + k);
        acc0 = WMMA_F16(a, b0, acc0);
        acc1 = WMMA_F16(a, b1, acc1);
    }

    // Publish gate tiles to LDS
    {
        int rb = (lane >> 4) * 8;
#pragma unroll
        for (int j = 0; j < 8; ++j) {
            lds[gate][rb + j][lr]      = acc0[j];
            lds[gate][rb + j][16 + lr] = acc1[j];
        }
    }
    __syncthreads();

    // Cell update: 512 elements / 128 threads = 4 each
#pragma unroll
    for (int e = 0; e < 4; ++e) {
        int idx = threadIdx.x * 4 + e;     // 0..511
        int r = idx >> 5, c = idx & 31;
        float gi = lds[0][r][c];
        float gf = lds[1][r][c];
        float gg = lds[2][r][c];
        float go = lds[3][r][c];
        float si = 1.0f / (1.0f + __expf(-gi));
        float sf = 1.0f / (1.0f + __expf(-gf));
        float tg = tanhf(gg);
        float so = 1.0f / (1.0f + __expf(-go));
        size_t off = (size_t)(m0 + r) * H_ + n0 + c;
        float cn = sf * cbuf[off] + si * tg;
        cbuf[off] = cn;
        float hn = so * tanhf(cn);
        hout[off] = (_Float16)hn;
        if (hseq)
            hseq[((size_t)(m0 + r) * T_ + t) * H_ + n0 + c] = (_Float16)hn;
    }
}

// ---------------------------------------------------------------------------
// FC head: out[M,N] f32 = A[M,K] f16 @ W[N,K]^T + bias  (tiny: 64 tiles)
// ---------------------------------------------------------------------------
__global__ void gemm_fc_kernel(const _Float16* __restrict__ A,
                               const _Float16* __restrict__ W,
                               const float* __restrict__ bias,
                               float* __restrict__ out,
                               int M, int N, int K) {
    int wave = blockIdx.x * (blockDim.x >> 5) + (threadIdx.x >> 5);
    int lane = threadIdx.x & 31;
    int tn   = N >> 4;
    int m0   = (wave / tn) << 4;
    int n0   = (wave % tn) << 4;
    int lr   = lane & 15;
    int kh   = (lane >> 4) << 3;

    const _Float16* ap = A + (size_t)(m0 + lr) * K + kh;
    const _Float16* bp = W + (size_t)(n0 + lr) * K + kh;

    v8f acc = {};
    for (int k = 0; k < K; k += 32) {
        v16h a = load_frag(ap + k);
        v16h b = load_frag(bp + k);
        acc = WMMA_F16(a, b, acc);
    }

    int col   = n0 + lr;
    int rbase = m0 + (lane >> 4) * 8;
    float b0  = bias[col];
#pragma unroll
    for (int j = 0; j < 8; ++j)
        out[(size_t)(rbase + j) * N + col] = acc[j] + b0;
}

// ---------------------------------------------------------------------------
extern "C" void kernel_launch(void* const* d_in, const int* in_sizes, int n_in,
                              void* d_out, int out_size, void* d_ws, size_t ws_size,
                              hipStream_t stream) {
    (void)in_sizes; (void)n_in; (void)out_size; (void)ws_size;

    const float* x    = (const float*)d_in[0];
    const float* Wih0 = (const float*)d_in[1];
    const float* Whh0 = (const float*)d_in[2];
    const float* bih0 = (const float*)d_in[3];
    const float* bhh0 = (const float*)d_in[4];
    const float* Wih1 = (const float*)d_in[5];
    const float* Whh1 = (const float*)d_in[6];
    const float* bih1 = (const float*)d_in[7];
    const float* bhh1 = (const float*)d_in[8];
    const float* fcw  = (const float*)d_in[9];
    const float* fcb  = (const float*)d_in[10];
    float* out = (float*)d_out;

    // Workspace carve-out
    char* ws = (char*)d_ws;
    size_t off = 0;
    auto carve = [&](size_t bytes) -> char* {
        char* p = ws + off;
        off += (bytes + 255) & ~(size_t)255;
        return p;
    };
    _Float16* xp    = (_Float16*)carve((size_t)B_ * T_ * G4H * 2); // reused L0/L1
    _Float16* hseq  = (_Float16*)carve((size_t)B_ * T_ * H_  * 2);
    _Float16* xh    = (_Float16*)carve((size_t)B_ * T_ * I_  * 2);
    _Float16* wih0h = (_Float16*)carve((size_t)G4H * I_ * 2);
    _Float16* whh0h = (_Float16*)carve((size_t)G4H * H_ * 2);
    _Float16* wih1h = (_Float16*)carve((size_t)G4H * H_ * 2);
    _Float16* whh1h = (_Float16*)carve((size_t)G4H * H_ * 2);
    _Float16* fcwh  = (_Float16*)carve((size_t)O_  * H_ * 2);
    _Float16* hbuf0 = (_Float16*)carve((size_t)B_ * H_ * 2);
    _Float16* hbuf1 = (_Float16*)carve((size_t)B_ * H_ * 2);
    float*    cbuf  = (float*)   carve((size_t)B_ * H_ * 4);

    auto cvt = [&](const float* s, _Float16* d, int n) {
        cvt_f16_kernel<<<(n + 255) / 256, 256, 0, stream>>>(s, d, n);
    };
    cvt(x,    xh,    B_ * T_ * I_);
    cvt(Wih0, wih0h, G4H * I_);
    cvt(Whh0, whh0h, G4H * H_);
    cvt(Wih1, wih1h, G4H * H_);
    cvt(Whh1, whh1h, G4H * H_);
    cvt(fcw,  fcwh,  O_ * H_);

    const int M = B_ * T_;

    // ---- Layer 0 ----
    {
        int waves = (M / 16) * (G4H / 64);   // 1x4 register blocking
        gemm_proj_kernel<<<waves / 4, 128, 0, stream>>>(
            xh, wih0h, bih0, bhh0, xp, M, G4H, I_);
    }
    init_state_kernel<<<(B_ * H_ + 255) / 256, 256, 0, stream>>>(hbuf0, cbuf, B_ * H_);
    for (int t = 0; t < T_; ++t) {
        const _Float16* hi = (t & 1) ? hbuf1 : hbuf0;
        _Float16*       ho = (t & 1) ? hbuf0 : hbuf1;
        lstm_step_kernel<<<(B_ / 16) * (H_ / 32), 128, 0, stream>>>(
            xp, whh0h, hi, ho, cbuf, hseq, t);
    }

    // ---- Layer 1 (reuse xp buffer; stream order guarantees layer-0 done) ----
    {
        int waves = (M / 16) * (G4H / 64);
        gemm_proj_kernel<<<waves / 4, 128, 0, stream>>>(
            hseq, wih1h, bih1, bhh1, xp, M, G4H, H_);
    }
    init_state_kernel<<<(B_ * H_ + 255) / 256, 256, 0, stream>>>(hbuf0, cbuf, B_ * H_);
    for (int t = 0; t < T_; ++t) {
        const _Float16* hi = (t & 1) ? hbuf1 : hbuf0;
        _Float16*       ho = (t & 1) ? hbuf0 : hbuf1;
        lstm_step_kernel<<<(B_ / 16) * (H_ / 32), 128, 0, stream>>>(
            xp, whh1h, hi, ho, cbuf, (_Float16*)nullptr, t);
    }

    // ---- FC head: final h is in hbuf0 (t=255 writes parity 0) ----
    {
        int waves = (B_ / 16) * (O_ / 16);   // 64 waves
        gemm_fc_kernel<<<waves / 4, 128, 0, stream>>>(
            hbuf0, fcwh, fcb, out, B_, O_, H_);
    }
}
